// CAMLayer_31284541784147
// MI455X (gfx1250) — matrix operand order repaired and checked by reading, not compiled, not run on hardware
//
#include <hip/hip_runtime.h>

#define B_    8
#define NS_   25
#define NQ_   75
#define WAY_  5
#define C_    512
#define HW_   36
#define MID_  6
#define NCLS_ 64
#define BJ_   (B_ * NQ_)      // 600
#define MROWS_ (WAY_ * HW_)   // 180

typedef __attribute__((ext_vector_type(16))) __bf16 v16bf;
typedef __attribute__((ext_vector_type(8)))  float  v8f;
typedef unsigned short u16t;
typedef unsigned int   u32t;

__device__ __forceinline__ u16t f2bf(float f) {
  union { float f; u32t u; } x; x.f = f;
  u32t u = x.u;
  u32t r = u + 0x7FFFu + ((u >> 16) & 1u);   // round-to-nearest-even
  return (u16t)(r >> 16);
}
__device__ __forceinline__ float bf2f(u16t h) {
  union { u32t u; float f; } x; x.u = ((u32t)h) << 16;
  return x.f;
}

union FragAB { v16bf v; u32t u[8]; };

// A row-major [nrows x K] bf16 -> 16x32 A fragment (ISA 7.12.2):
// lane L: M=L%16, half=L/16; VGPR v<4: K=half*8+2v+pos ; v>=4: K=16+half*8+2(v-4)+pos
__device__ __forceinline__ v16bf load_frag_a(const u16t* A, int row0, int nrows,
                                             int ldk, int kbase, int lane) {
  FragAB r;
  int m = lane & 15, half = lane >> 4;
  int row = row0 + m;
  if (row < nrows) {
    const u16t* p = A + (size_t)row * ldk + kbase + half * 8;
#pragma unroll
    for (int v = 0; v < 4; ++v) {
      r.u[v]     = *(const u32t*)(p + 2 * v);
      r.u[v + 4] = *(const u32t*)(p + 16 + 2 * v);
    }
  } else {
#pragma unroll
    for (int v = 0; v < 8; ++v) r.u[v] = 0u;
  }
  return r.v;
}

// B stored transposed Bt[N x K] row-major bf16 -> 32x16 B fragment:
// lane L: N=L%16, half=L/16; VGPR v<4: K=half*8+2v+pos ; v>=4: K=16+half*8+2(v-4)+pos
__device__ __forceinline__ v16bf load_frag_bt(const u16t* Bt, int col0, int ncols,
                                              int ldk, int kbase, int lane) {
  FragAB r;
  int n = lane & 15, half = lane >> 4;
  int col = col0 + n;
  if (col < ncols) {
    const u16t* p = Bt + (size_t)col * ldk + kbase + half * 8;
#pragma unroll
    for (int v = 0; v < 8; ++v) r.u[v] = *(const u32t*)(p + 2 * v + (v >= 4 ? 8 : 0));
  } else {
#pragma unroll
    for (int v = 0; v < 8; ++v) r.u[v] = 0u;
  }
  return r.v;
}

// ---------------- small prep kernels ----------------

__global__ void k_cvt_bf16(const float* __restrict__ src, u16t* __restrict__ dst, int n) {
  int i = blockIdx.x * blockDim.x + threadIdx.x;
  if (i < n) dst[i] = f2bf(src[i]);
}

// protos[b,w,c,x] (f32) and f1n[b, w*36+x, c] (bf16, L2-normalized over c per (b,w,x))
__global__ void k_protos(const float* __restrict__ sf, const float* __restrict__ st,
                         float* __restrict__ protos, u16t* __restrict__ f1n) {
  int bw = blockIdx.x;                 // b*WAY + w
  int b = bw / WAY_, w = bw % WAY_;
  extern __shared__ float sh[];
  float* sp = sh;                      // [C_*HW_], indexed d = c*36 + x
  float* sinvn = sh + C_ * HW_;        // [HW_]
  __shared__ float slt[NS_];
  int t = threadIdx.x;                 // 512 threads
  if (t < NS_) slt[t] = st[((size_t)b * NS_ + t) * WAY_ + w];
  __syncthreads();
  float cnt = 0.f;
  for (int n = 0; n < NS_; ++n) cnt += slt[n];
  float icnt = 1.f / cnt;
  for (int d = t; d < C_ * HW_; d += 512) {
    float acc = 0.f;
    for (int n = 0; n < NS_; ++n)
      acc += slt[n] * sf[((size_t)(b * NS_ + n)) * (C_ * HW_) + d];
    acc *= icnt;
    sp[d] = acc;
    protos[(size_t)bw * (C_ * HW_) + d] = acc;
  }
  __syncthreads();
  if (t < HW_) {
    float s = 0.f;
    for (int c = 0; c < C_; ++c) { float v = sp[c * HW_ + t]; s += v * v; }
    sinvn[t] = 1.f / fmaxf(sqrtf(s), 1e-12f);
  }
  __syncthreads();
  for (int e = t; e < C_ * HW_; e += 512) {
    int x = e >> 9, c = e & (C_ - 1);
    f1n[((size_t)b * MROWS_ + w * HW_ + x) * C_ + c] = f2bf(sp[c * HW_ + x] * sinvn[x]);
  }
}

// f2n[bj, y, c] bf16 (transposed, L2-normalized over c per (bj,y)); qn[bj,y] = max(||.||,eps)
__global__ void k_qnorm(const float* __restrict__ qf, u16t* __restrict__ f2n,
                        float* __restrict__ qn) {
  int bj = blockIdx.x;
  const float* Q = qf + (size_t)bj * (C_ * HW_);   // [c, y], y fastest
  __shared__ float sinv[HW_];
  int t = threadIdx.x;                 // 512
  if (t < HW_) {
    float s = 0.f;
    for (int c = 0; c < C_; ++c) { float v = Q[(size_t)c * HW_ + t]; s += v * v; }
    float n = fmaxf(sqrtf(s), 1e-12f);
    sinv[t] = 1.f / n;
    qn[(size_t)bj * HW_ + t] = n;
  }
  __syncthreads();
  for (int e = t; e < C_ * HW_; e += 512) {
    int y = e >> 9, c = e & (C_ - 1);
    f2n[(size_t)bj * (HW_ * C_) + (size_t)y * C_ + c] = f2bf(Q[(size_t)c * HW_ + y] * sinv[y]);
  }
}

// ---------------- WMMA GEMM: a1[bj, i*36+x, y] = f1n[b] @ f2n[bj]^T ----------------

__global__ void k_a1_gemm(const u16t* __restrict__ f1n, const u16t* __restrict__ f2n,
                          float* __restrict__ a1) {
  int bj = blockIdx.x;
  int tile = blockIdx.y;               // 12 M-tiles x 3 N-tiles
  int mo = (tile % 12) * 16, no = (tile / 12) * 16;
  int lane = threadIdx.x;
  int b = bj / NQ_;
  const u16t* A  = f1n + (size_t)b * MROWS_ * C_;
  const u16t* Bt = f2n + (size_t)bj * HW_ * C_;
  v8f acc = {0.f, 0.f, 0.f, 0.f, 0.f, 0.f, 0.f, 0.f};
  for (int k = 0; k < C_; k += 32) {
    v16bf fa = load_frag_a(A, mo, MROWS_, C_, k, lane);
    v16bf fb = load_frag_bt(Bt, no, HW_, C_, k, lane);
    acc = __builtin_amdgcn_wmma_f32_16x16x32_bf16(false, fa, false, fb,
                                                  (short)0, acc, false, false);
  }
  int col  = no + (lane & 15);
  int rowb = mo + (lane >> 4) * 8;
  if (col < HW_) {
    float* out = a1 + (size_t)bj * MROWS_ * HW_;
#pragma unroll
    for (int r = 0; r < 8; ++r) {
      int row = rowb + r;
      if (row < MROWS_) out[(size_t)row * HW_ + col] = acc[r];
    }
  }
}

// ---------------- attention small ops ----------------

// m1[b,i,j,y] = mean_x a1 ; m2[b,i,j,x] = mean_y a1
__global__ void k_means(const float* __restrict__ a1, float* __restrict__ m1,
                        float* __restrict__ m2) {
  int id = blockIdx.x;                 // bj*WAY + i
  int bj = id / WAY_, i = id % WAY_;
  int b = bj / NQ_, j = bj % NQ_;
  const float* A = a1 + ((size_t)bj * MROWS_ + i * HW_) * HW_;
  int bij = (b * WAY_ + i) * NQ_ + j;
  int t = threadIdx.x;
  if (t < HW_) {
    float s = 0.f;
    for (int x = 0; x < HW_; ++x) s += A[(size_t)x * HW_ + t];
    m1[(size_t)bij * HW_ + t] = s * (1.f / HW_);
  } else if (t >= 64 && t < 64 + HW_) {
    int x = t - 64;
    float s = 0.f;
    for (int y = 0; y < HW_; ++y) s += A[(size_t)x * HW_ + y];
    m2[(size_t)bij * HW_ + x] = s * (1.f / HW_);
  }
}

// conv1 -> BN(batch stats over all b,j,i) -> ReLU -> conv2; one block per attention path
__global__ void k_mlp(const float* __restrict__ m1, const float* __restrict__ m2,
                      const float* __restrict__ w1, const float* __restrict__ b1,
                      const float* __restrict__ g,  const float* __restrict__ beta,
                      const float* __restrict__ w2, const float* __restrict__ b2,
                      float* __restrict__ p1, float* __restrict__ p2) {
  const float* m = (blockIdx.x == 0) ? m1 : m2;
  float* p       = (blockIdx.x == 0) ? p1 : p2;
  __shared__ float sw1[MID_ * HW_], sw2[HW_ * MID_];
  __shared__ float sb1[MID_], sb2c[HW_];
  __shared__ float ssum[MID_], ssq[MID_], sA[MID_], sB[MID_];
  int t = threadIdx.x;
  for (int e = t; e < MID_ * HW_; e += blockDim.x) sw1[e] = w1[e];
  for (int e = t; e < HW_ * MID_; e += blockDim.x) sw2[e] = w2[e];
  if (t < MID_) { sb1[t] = b1[t]; ssum[t] = 0.f; ssq[t] = 0.f; }
  if (t < HW_) sb2c[t] = b2[t];
  __syncthreads();
  const int P = B_ * NQ_ * WAY_;       // 3000 positions
  float psum[MID_], psq[MID_];
#pragma unroll
  for (int c = 0; c < MID_; ++c) { psum[c] = 0.f; psq[c] = 0.f; }
  for (int pos = t; pos < P; pos += blockDim.x) {
    int b = pos / (NQ_ * WAY_);
    int rem = pos % (NQ_ * WAY_);
    int j = rem / WAY_, i = rem % WAY_;
    const float* mm = m + (((size_t)(b * WAY_ + i)) * NQ_ + j) * HW_;
#pragma unroll
    for (int c = 0; c < MID_; ++c) {
      float y = sb1[c];
      for (int k = 0; k < HW_; ++k) y += sw1[c * HW_ + k] * mm[k];
      psum[c] += y; psq[c] += y * y;
    }
  }
#pragma unroll
  for (int c = 0; c < MID_; ++c) { atomicAdd(&ssum[c], psum[c]); atomicAdd(&ssq[c], psq[c]); }
  __syncthreads();
  if (t < MID_) {
    float mu  = ssum[t] / (float)P;
    float var = ssq[t] / (float)P - mu * mu;
    float inv = rsqrtf(var + 1e-5f) * g[t];
    sA[t] = inv;
    sB[t] = beta[t] - mu * inv;
  }
  __syncthreads();
  for (int pos = t; pos < P; pos += blockDim.x) {
    int b = pos / (NQ_ * WAY_);
    int rem = pos % (NQ_ * WAY_);
    int j = rem / WAY_, i = rem % WAY_;
    const float* mm = m + (((size_t)(b * WAY_ + i)) * NQ_ + j) * HW_;
    float z[MID_];
#pragma unroll
    for (int c = 0; c < MID_; ++c) {
      float y = sb1[c];
      for (int k = 0; k < HW_; ++k) y += sw1[c * HW_ + k] * mm[k];
      float zz = sA[c] * y + sB[c];
      z[c] = zz > 0.f ? zz : 0.f;
    }
    float* outp = p + (((size_t)(b * WAY_ + i)) * NQ_ + j) * HW_;
    for (int k = 0; k < HW_; ++k) {
      float o = sb2c[k];
#pragma unroll
      for (int c = 0; c < MID_; ++c) o += sw2[k * MID_ + c] * z[c];
      outp[k] = o;
    }
  }
}

// s = mean(a*att_pre) ; att = softmax(s/0.025)+1 (both directions)
__global__ void k_soft(const float* __restrict__ a1, const float* __restrict__ p1,
                       const float* __restrict__ p2, float* __restrict__ att1,
                       float* __restrict__ att2) {
  int id = blockIdx.x;                 // bj*WAY + i
  int bj = id / WAY_, i = id % WAY_;
  int b = bj / NQ_, j = bj % NQ_;
  const float* A = a1 + ((size_t)bj * MROWS_ + i * HW_) * HW_;
  int bij = (b * WAY_ + i) * NQ_ + j;
  __shared__ float sa[HW_ * HW_];
  __shared__ float sp1[HW_], sp2[HW_], s1[HW_], s2[HW_];
  int t = threadIdx.x;                 // 128
  for (int e = t; e < HW_ * HW_; e += 128) sa[e] = A[e];
  if (t < HW_) { sp1[t] = p1[(size_t)bij * HW_ + t]; sp2[t] = p2[(size_t)bij * HW_ + t]; }
  __syncthreads();
  if (t < HW_) {
    float s = 0.f;
    for (int y = 0; y < HW_; ++y) s += sa[t * HW_ + y] * sp1[y];
    s1[t] = s * (1.f / HW_);
  } else if (t >= 64 && t < 64 + HW_) {
    int y = t - 64;
    float s = 0.f;
    for (int x = 0; x < HW_; ++x) s += sa[x * HW_ + y] * sp2[x];
    s2[y] = s * (1.f / HW_);
  }
  __syncthreads();
  if (t == 0) {
    float mx = s1[0];
    for (int x = 1; x < HW_; ++x) mx = fmaxf(mx, s1[x]);
    float e[HW_], sum = 0.f;
    for (int x = 0; x < HW_; ++x) { e[x] = __expf((s1[x] - mx) * 40.f); sum += e[x]; }
    float inv = 1.f / sum;
    for (int x = 0; x < HW_; ++x) att1[(size_t)bij * HW_ + x] = e[x] * inv + 1.f;
  }
  if (t == 64) {
    float mx = s2[0];
    for (int y = 1; y < HW_; ++y) mx = fmaxf(mx, s2[y]);
    float e[HW_], sum = 0.f;
    for (int y = 0; y < HW_; ++y) { e[y] = __expf((s2[y] - mx) * 40.f); sum += e[y]; }
    float inv = 1.f / sum;
    for (int y = 0; y < HW_; ++y) att2[(size_t)bij * HW_ + y] = e[y] * inv + 1.f;
  }
}

// proto_pooled = mean_x protos*att1 ; proto_norm = l2norm over c -> pnorm[(bj*WAY+i), c]
__global__ void k_pool(const float* __restrict__ protos, const float* __restrict__ att1,
                       float* __restrict__ pnorm) {
  int id = blockIdx.x;                 // bj*WAY + i
  int bj = id / WAY_, i = id % WAY_;
  int b = bj / NQ_, j = bj % NQ_;
  __shared__ float sat[HW_];
  __shared__ float sred[8];
  int t = threadIdx.x;                 // 256
  int bij = (b * WAY_ + i) * NQ_ + j;
  if (t < HW_) sat[t] = att1[(size_t)bij * HW_ + t];
  __syncthreads();
  const float* P = protos + ((size_t)(b * WAY_ + i)) * (C_ * HW_);
  float pp[2];
  float ss = 0.f;
#pragma unroll
  for (int r = 0; r < 2; ++r) {
    int c = t + r * 256;
    float acc = 0.f;
    for (int x = 0; x < HW_; ++x) acc += P[(size_t)c * HW_ + x] * sat[x];
    acc *= (1.f / HW_);
    pp[r] = acc;
    ss += acc * acc;
  }
  int lane = t & 31, wave = t >> 5;
  for (int off = 16; off > 0; off >>= 1) ss += __shfl_down(ss, off, 32);
  if (lane == 0) sred[wave] = ss;
  __syncthreads();
  if (t == 0) {
    float tot = 0.f;
    for (int w = 0; w < 8; ++w) tot += sred[w];
    sred[0] = 1.f / fmaxf(sqrtf(tot), 1e-12f);
  }
  __syncthreads();
  float inv = sred[0];
#pragma unroll
  for (int r = 0; r < 2; ++r) pnorm[(size_t)id * C_ + t + r * 256] = pp[r] * inv;
}

// cls_scores[(bj), i, y] = 7 * sum_c f2n[bj,y,c] * pnorm[(bj,i),c]
__global__ void k_cls(const u16t* __restrict__ f2n, const float* __restrict__ pnorm,
                      float* __restrict__ out2) {
  int id = blockIdx.x;                 // bj*WAY + i
  int bj = id / WAY_;
  const u16t* F = f2n + (size_t)bj * (HW_ * C_);
  const float* P = pnorm + (size_t)id * C_;
  int lane = threadIdx.x & 31, wave = threadIdx.x >> 5;
  for (int y = wave; y < HW_; y += 8) {
    float acc = 0.f;
    for (int c = lane; c < C_; c += 32) acc += bf2f(F[(size_t)y * C_ + c]) * P[c];
    for (int off = 16; off > 0; off >>= 1) acc += __shfl_down(acc, off, 32);
    if (lane == 0) out2[(size_t)id * HW_ + y] = 7.0f * acc;
  }
}

// logits[(bj), o, y] = att2sel[bj,y]*qn[bj,y]*(cls_w @ f2n[bj])[o,y] + cls_b[o]
__global__ void k_logits(const u16t* __restrict__ wcls, const u16t* __restrict__ f2n,
                         const float* __restrict__ qn, const float* __restrict__ att2,
                         const float* __restrict__ qt, const float* __restrict__ clsb,
                         float* __restrict__ out) {
  int bj = blockIdx.x;
  int tile = blockIdx.y;               // 4 M-tiles x 3 N-tiles
  int mo = (tile % 4) * 16, no = (tile / 4) * 16;
  int lane = threadIdx.x;
  int b = bj / NQ_, j = bj % NQ_;
  const u16t* Bt = f2n + (size_t)bj * (HW_ * C_);
  v8f acc = {0.f, 0.f, 0.f, 0.f, 0.f, 0.f, 0.f, 0.f};
  for (int k = 0; k < C_; k += 32) {
    v16bf fa = load_frag_a(wcls, mo, NCLS_, C_, k, lane);
    v16bf fb = load_frag_bt(Bt, no, HW_, C_, k, lane);
    acc = __builtin_amdgcn_wmma_f32_16x16x32_bf16(false, fa, false, fb,
                                                  (short)0, acc, false, false);
  }
  int col = no + (lane & 15);
  if (col < HW_) {
    float sel = 0.f;
#pragma unroll
    for (int i = 0; i < WAY_; ++i)
      sel += qt[(size_t)bj * WAY_ + i] *
             att2[(((size_t)(b * WAY_ + i)) * NQ_ + j) * HW_ + col];
    float scale = sel * qn[(size_t)bj * HW_ + col];
    int rowb = mo + (lane >> 4) * 8;
#pragma unroll
    for (int r = 0; r < 8; ++r) {
      int o = rowb + r;
      out[((size_t)bj * NCLS_ + o) * HW_ + col] = scale * acc[r] + clsb[o];
    }
  }
}

// ---------------- launch ----------------

extern "C" void kernel_launch(void* const* d_in, const int* in_sizes, int n_in,
                              void* d_out, int out_size, void* d_ws, size_t ws_size,
                              hipStream_t stream) {
  (void)in_sizes; (void)n_in; (void)out_size; (void)ws_size;
  const float* support_feat = (const float*)d_in[0];
  const float* query_feat   = (const float*)d_in[1];
  const float* support_tg   = (const float*)d_in[2];
  const float* query_tg     = (const float*)d_in[3];
  const float* conv1_w = (const float*)d_in[4];
  const float* conv1_b = (const float*)d_in[5];
  const float* bn_g    = (const float*)d_in[6];
  const float* bn_b    = (const float*)d_in[7];
  const float* conv2_w = (const float*)d_in[8];
  const float* conv2_b = (const float*)d_in[9];
  const float* cls_w   = (const float*)d_in[10];
  const float* cls_b   = (const float*)d_in[11];

  char* w = (char*)d_ws;
  auto take = [&](size_t bytes) {
    char* p = w;
    w += (bytes + 255) & ~(size_t)255;
    return (void*)p;
  };
  float* protos = (float*)take((size_t)B_ * WAY_ * C_ * HW_ * 4);
  u16t*  f1n    = (u16t*) take((size_t)B_ * MROWS_ * C_ * 2);
  u16t*  f2n    = (u16t*) take((size_t)BJ_ * HW_ * C_ * 2);
  float* qn     = (float*)take((size_t)BJ_ * HW_ * 4);
  float* a1     = (float*)take((size_t)BJ_ * MROWS_ * HW_ * 4);
  float* m1     = (float*)take((size_t)B_ * WAY_ * NQ_ * HW_ * 4);
  float* m2     = (float*)take((size_t)B_ * WAY_ * NQ_ * HW_ * 4);
  float* p1     = (float*)take((size_t)B_ * WAY_ * NQ_ * HW_ * 4);
  float* p2     = (float*)take((size_t)B_ * WAY_ * NQ_ * HW_ * 4);
  float* att1   = (float*)take((size_t)B_ * WAY_ * NQ_ * HW_ * 4);
  float* att2   = (float*)take((size_t)B_ * WAY_ * NQ_ * HW_ * 4);
  float* pnorm  = (float*)take((size_t)BJ_ * WAY_ * C_ * 4);
  u16t*  wcls   = (u16t*) take((size_t)NCLS_ * C_ * 2);

  float* out_logits = (float*)d_out;
  float* out_cls    = out_logits + (size_t)BJ_ * NCLS_ * HW_;

  k_cvt_bf16<<<dim3((NCLS_ * C_ + 255) / 256), dim3(256), 0, stream>>>(cls_w, wcls, NCLS_ * C_);

  size_t shp = (size_t)(C_ * HW_ + 64) * sizeof(float);
  k_protos<<<dim3(B_ * WAY_), dim3(512), shp, stream>>>(support_feat, support_tg, protos, f1n);
  k_qnorm<<<dim3(BJ_), dim3(512), 0, stream>>>(query_feat, f2n, qn);

  k_a1_gemm<<<dim3(BJ_, 36), dim3(32), 0, stream>>>(f1n, f2n, a1);

  k_means<<<dim3(BJ_ * WAY_), dim3(128), 0, stream>>>(a1, m1, m2);
  k_mlp<<<dim3(2), dim3(1024), 0, stream>>>(m1, m2, conv1_w, conv1_b, bn_g, bn_b,
                                            conv2_w, conv2_b, p1, p2);
  k_soft<<<dim3(BJ_ * WAY_), dim3(128), 0, stream>>>(a1, p1, p2, att1, att2);

  k_pool<<<dim3(BJ_ * WAY_), dim3(256), 0, stream>>>(protos, att1, pnorm);
  k_cls<<<dim3(BJ_ * WAY_), dim3(256), 0, stream>>>(f2n, pnorm, out_cls);
  k_logits<<<dim3(BJ_, 12), dim3(32), 0, stream>>>(wcls, f2n, qn, att2, query_tg,
                                                   cls_b, out_logits);
}
